// HomeWorld_11871289606540
// MI455X (gfx1250) — compile-verified, matrix operand order checked by reading, not run on hardware
//
#include <hip/hip_runtime.h>

// ---------------------------------------------------------------------------
// MI455X (gfx1250, wave32) LSTM-DQN forward.
//   B=16384, T=30, V=100, E=20, H=256, A=40
// Strategy: f16 WMMA (v_wmma_f32_16x16x32_f16). Each wave owns a 32-row batch
// tile (two 16-row WMMA A tiles) for the whole sequence: h in LDS f16, loaded as
// A-fragments, c in LDS f32, weights pre-packed into per-lane fragment order
// (one contiguous v16h per lane per fragment). B fragments are double-buffered
// so WMMA never waits on the load issued in the same K-step, and each B
// fragment is reused by both row tiles (halves L2 weight traffic).
// ---------------------------------------------------------------------------

typedef _Float16 f16;
typedef __attribute__((ext_vector_type(16))) _Float16 v16h;
typedef __attribute__((ext_vector_type(8)))  float    v8f;

union F16x16 { v16h v; unsigned u[8]; f16 h[16]; };

// 16-bit WMMA A/B fragment K index for half-element h (0..15) in lane-group g
// (g = lane>>4).  Derived from ISA 7.12.2 "16-bit A-Matrix 16x32" table:
//   VGPR0..3 : K = {0,1},{2,3},{4,5},{6,7}  (+8 for lanes 16..31)
//   VGPR4..7 : K = 16..23                   (+8 for lanes 16..31)
__device__ __host__ inline int kform(int h, int g) {
    return ((h >> 3) << 4) + (((h >> 1) & 3) << 1) + (h & 1) + (g << 3);
}

__device__ inline float sigm_(float x) { return 1.0f / (1.0f + __expf(-x)); }
__device__ inline float tanh_(float x) {
    float t = __expf(-2.0f * fabsf(x));
    float r = (1.0f - t) / (1.0f + t);
    return copysignf(r, x);
}

// ---------------------------------------------------------------------------
// Workspace layout (all offsets 256B aligned)
// ---------------------------------------------------------------------------
constexpr size_t OFF_EMB = 0;                    // 100*32 f16            = 6400 B
constexpr size_t OFF_QW  = 16384;                // 9*64*512 f16          = 589824 B
constexpr size_t OFF_OW  = OFF_QW + 589824;      // 606208
constexpr size_t OFF_FC1 = OFF_OW + 589824;      // 1196032 ; 16*7*512 f16 = 114688 B
constexpr size_t OFF_FC2 = OFF_FC1 + 114688;     // 1310720 ; 4*3*512 f16  = 12288 B
constexpr size_t OFF_EQ  = OFF_FC2 + 12288;      // 1323008 ; 16384*256 f16 = 8388608 B
constexpr size_t OFF_EO  = OFF_EQ + 8388608;     // 9711616
// total ~18.1 MB

// ---------------------------------------------------------------------------
// Pack kernels: scatter f32 weights into per-lane v16h fragment order.
// Fragment id = kt*NT + nt ; within fragment: lane 0..31, 16 halves.
// B-matrix (KxN): N = nt*16 + (lane&15), K = kt*32 + kform(h, lane>>4).
// ---------------------------------------------------------------------------
__global__ void pack_emb_kernel(const float* __restrict__ emb, f16* __restrict__ dst) {
    int tid = blockIdx.x * blockDim.x + threadIdx.x;
    if (tid >= 100 * 32) return;
    int v = tid >> 5, k = tid & 31;
    dst[tid] = (k < 20) ? (f16)emb[v * 20 + k] : (f16)0.0f;
}

// Packs [Wih | Whh] (rows = gate output n in 0..1023) as K=288 (x padded to 32).
__global__ void pack_w_kernel(const float* __restrict__ Wih,
                              const float* __restrict__ Whh,
                              f16* __restrict__ dst) {
    int tid = blockIdx.x * blockDim.x + threadIdx.x;
    if (tid >= 9 * 64 * 32) return;
    int lane = tid & 31, frag = tid >> 5;
    int nt = frag % 64, kt = frag / 64;
    int g = lane >> 4, nl = lane & 15;
    int n = nt * 16 + nl;
    F16x16 o;
#pragma unroll
    for (int h = 0; h < 16; ++h) {
        int k = kt * 32 + kform(h, g);
        float w;
        if (kt == 0) w = (k < 20) ? Wih[n * 20 + k] : 0.0f;   // x part, padded
        else         w = Whh[n * 256 + (k - 32)];             // h part
        o.h[h] = (f16)w;
    }
    ((v16h*)dst)[frag * 32 + lane] = o.v;
}

__global__ void pack_fc1_kernel(const float* __restrict__ W, f16* __restrict__ dst) {
    int tid = blockIdx.x * blockDim.x + threadIdx.x;
    if (tid >= 16 * 7 * 32) return;
    int lane = tid & 31, frag = tid >> 5;
    int nt = frag % 7, kt = frag / 7;
    int g = lane >> 4, nl = lane & 15;
    int n = nt * 16 + nl;
    F16x16 o;
#pragma unroll
    for (int h = 0; h < 16; ++h) {
        int k = kt * 32 + kform(h, g);                        // k in 0..511
        o.h[h] = (n < 100) ? (f16)W[n * 512 + k] : (f16)0.0f;
    }
    ((v16h*)dst)[frag * 32 + lane] = o.v;
}

__global__ void pack_fc2_kernel(const float* __restrict__ W, f16* __restrict__ dst) {
    int tid = blockIdx.x * blockDim.x + threadIdx.x;
    if (tid >= 4 * 3 * 32) return;
    int lane = tid & 31, frag = tid >> 5;
    int nt = frag % 3, kt = frag / 3;
    int g = lane >> 4, nl = lane & 15;
    int n = nt * 16 + nl;
    F16x16 o;
#pragma unroll
    for (int h = 0; h < 16; ++h) {
        int k = kt * 32 + kform(h, g);                        // K padded 100->128
        o.h[h] = (k < 100 && n < 40) ? (f16)W[n * 100 + k] : (f16)0.0f;
    }
    ((v16h*)dst)[frag * 32 + lane] = o.v;
}

// ---------------------------------------------------------------------------
// LSTM encoder. grid = (B/32, 2), block = 32 (one wave, 32 rows = 2 A-tiles).
// LDS per block: cS 32KB (f32) + hS 16KB (f16) = 48KB.
// Per K-step: 8 B128 loads (4 gate fragments, double-buffered) feed 8 WMMAs
// (4 gates x 2 row tiles) -> each load's latency is covered by a full K-step
// of matrix work plus the next-step loads already in flight.
// ---------------------------------------------------------------------------
__global__ __launch_bounds__(32) void lstm_enc_kernel(
    const int* __restrict__ input, const f16* __restrict__ emb16,
    const v16h* __restrict__ qW, const float* __restrict__ qb,
    const v16h* __restrict__ oW, const float* __restrict__ ob,
    f16* __restrict__ encq, f16* __restrict__ enco) {
    const int e = blockIdx.y;
    const v16h*  W    = e ? oW : qW;
    const float* bias = e ? ob : qb;
    f16*         encG = e ? enco : encq;

    const int lane = threadIdx.x & 31;
    const int g = lane >> 4, nl = lane & 15;
    const int rowBase = blockIdx.x * 32;

    __shared__ __align__(16) float cS[32][256];
    __shared__ __align__(16) f16   hS[32][256];

    // zero c (f32) and h (f16); zero our enc rows in global
    {
        float4 fz = {0.f, 0.f, 0.f, 0.f};
        uint4  uz = {0u, 0u, 0u, 0u};
        float4* cz = (float4*)&cS[0][0];
#pragma unroll 4
        for (int i = lane; i < 2048; i += 32) cz[i] = fz;
        uint4* hz = (uint4*)&hS[0][0];
#pragma unroll 4
        for (int i = lane; i < 1024; i += 32) hz[i] = uz;
        uint4* ez = (uint4*)(encG + (size_t)rowBase * 256);
#pragma unroll 4
        for (int i = lane; i < 1024; i += 32) ez[i] = uz;
    }

    for (int t = 0; t < 30; ++t) {
        // tokens for this lane's two rows (lanes 16..31 duplicate lanes 0..15)
        const int tokA = input[(rowBase + nl) * 60 + e * 30 + t];
        const int tokB = input[(rowBase + 16 + nl) * 60 + e * 30 + t];

        // x A-fragments: gather half-pairs from padded f16 embedding table
        F16x16 xfA, xfB;
#pragma unroll
        for (int j = 0; j < 8; ++j) {
            xfA.u[j] = *(const unsigned*)(emb16 + tokA * 32 + kform(2 * j, g));
            xfB.u[j] = *(const unsigned*)(emb16 + tokB * 32 + kform(2 * j, g));
        }

        // h A-fragments from LDS (previous step's h, row-major f16)
        F16x16 hfA[8], hfB[8];
#pragma unroll
        for (int kt = 0; kt < 8; ++kt)
#pragma unroll
            for (int j = 0; j < 8; ++j) {
                hfA[kt].u[j] = *(const unsigned*)&hS[nl][kt * 32 + kform(2 * j, g)];
                hfB[kt].u[j] = *(const unsigned*)&hS[16 + nl][kt * 32 + kform(2 * j, g)];
            }

        for (int n0 = 0; n0 < 16; ++n0) {
            if (n0 < 15)  // warm next n0's weight stream (global_prefetch_b8)
                __builtin_prefetch((const void*)(W + ((size_t)(n0 + 1) * 32 + lane)), 0, 0);

            // B fragments double-buffered across the K loop
            F16x16 bb[2][4];
#pragma unroll
            for (int q = 0; q < 4; ++q)
                bb[0][q].v = W[(size_t)(q * 16 + n0) * 32 + lane];  // kt = 0

            v8f accA[4], accB[4];
#pragma unroll
            for (int q = 0; q < 4; ++q) {
                accA[q] = (v8f){0.f, 0.f, 0.f, 0.f, 0.f, 0.f, 0.f, 0.f};
                accB[q] = (v8f){0.f, 0.f, 0.f, 0.f, 0.f, 0.f, 0.f, 0.f};
            }

#pragma unroll
            for (int kt = 0; kt < 9; ++kt) {
                const int cur = kt & 1, nxt = cur ^ 1;
                if (kt < 8) {  // issue next K-step's loads before this step's math
#pragma unroll
                    for (int q = 0; q < 4; ++q)
                        bb[nxt][q].v =
                            W[(size_t)(kt + 1) * 2048 + (size_t)(q * 16 + n0) * 32 + lane];
                }
                v16h aA = (kt == 0) ? xfA.v : hfA[kt - 1].v;
                v16h aB = (kt == 0) ? xfB.v : hfB[kt - 1].v;
#pragma unroll
                for (int q = 0; q < 4; ++q) {
                    accA[q] = __builtin_amdgcn_wmma_f32_16x16x32_f16(
                                  false, aA, false, bb[cur][q].v, (short)0, accA[q],
                                  false, false);
                    accB[q] = __builtin_amdgcn_wmma_f32_16x16x32_f16(
                                  false, aB, false, bb[cur][q].v, (short)0, accB[q],
                                  false, false);
                }
            }

            const int col = n0 * 16 + nl;
            const float bi = bias[col];
            const float bf = bias[256 + col];
            const float bg = bias[512 + col];
            const float bo = bias[768 + col];
#pragma unroll
            for (int r = 0; r < 8; ++r) {
                const int rl = g * 8 + r;  // C/D layout: M = r + 8*(lane>=16)
                {   // row tile A
                    float iv = accA[0][r] + bi;
                    float fv = accA[1][r] + bf;
                    float gv = accA[2][r] + bg;
                    float ov = accA[3][r] + bo;
                    float co = cS[rl][col];
                    float cn = sigm_(fv) * co + sigm_(iv) * tanh_(gv);
                    cS[rl][col] = cn;
                    hS[rl][col] = (f16)(sigm_(ov) * tanh_(cn));
                }
                {   // row tile B
                    float iv = accB[0][r] + bi;
                    float fv = accB[1][r] + bf;
                    float gv = accB[2][r] + bg;
                    float ov = accB[3][r] + bo;
                    float co = cS[16 + rl][col];
                    float cn = sigm_(fv) * co + sigm_(iv) * tanh_(gv);
                    cS[16 + rl][col] = cn;
                    hS[16 + rl][col] = (f16)(sigm_(ov) * tanh_(cn));
                }
            }
        }

        // enc = (1-m)*h + m*enc  ==  write h row when token != 0
        if (tokA != 0) {
            const uint4* src = (const uint4*)&hS[nl][g * 128];
            uint4* dst = (uint4*)(encG + ((size_t)(rowBase + nl)) * 256 + g * 128);
#pragma unroll
            for (int i = 0; i < 16; ++i) dst[i] = src[i];
        }
        if (tokB != 0) {
            const uint4* src = (const uint4*)&hS[16 + nl][g * 128];
            uint4* dst = (uint4*)(encG + ((size_t)(rowBase + 16 + nl)) * 256 + g * 128);
#pragma unroll
            for (int i = 0; i < 16; ++i) dst[i] = src[i];
        }
    }
}

// ---------------------------------------------------------------------------
// Head: relu(concat(q,o) @ fc1.T + b1) @ fc2.T + b2.
// grid = B/32, block = 64 (2 waves, 16 rows per wave). K1=512, N1 pad 112;
// K2 pad 128, N2 pad 48.
// ---------------------------------------------------------------------------
__global__ __launch_bounds__(64) void head_kernel(
    const f16* __restrict__ encq, const f16* __restrict__ enco,
    const v16h* __restrict__ W1, const float* __restrict__ b1,
    const v16h* __restrict__ W2, const float* __restrict__ b2,
    float* __restrict__ out) {
    const int wave = threadIdx.x >> 5;
    const int lane = threadIdx.x & 31;
    const int g = lane >> 4, nl = lane & 15;
    const int rowBase = blockIdx.x * 32 + wave * 16;

    __shared__ __align__(16) f16 xS[2][16][128];
    {
        uint4 uz = {0u, 0u, 0u, 0u};
        uint4* xz = (uint4*)&xS[wave][0][0];
#pragma unroll 4
        for (int i = lane; i < 256; i += 32) xz[i] = uz;
    }

    // A-fragments for layer 1: K = [encq(256) | enco(256)]
    F16x16 af[16];
#pragma unroll
    for (int kt = 0; kt < 16; ++kt) {
        const f16* src = (kt < 8 ? encq : enco) +
                         ((size_t)(rowBase + nl)) * 256 + (kt & 7) * 32;
#pragma unroll
        for (int j = 0; j < 8; ++j)
            af[kt].u[j] = *(const unsigned*)(src + kform(2 * j, g));
    }

    for (int nt = 0; nt < 7; ++nt) {
        v8f acc = {0.f, 0.f, 0.f, 0.f, 0.f, 0.f, 0.f, 0.f};
#pragma unroll
        for (int kt = 0; kt < 16; ++kt)
            acc = __builtin_amdgcn_wmma_f32_16x16x32_f16(
                      false, af[kt].v, false, W1[(size_t)(kt * 7 + nt) * 32 + lane],
                      (short)0, acc, false, false);
        const int col = nt * 16 + nl;
        const float bb = (col < 100) ? b1[col] : 0.0f;  // padded cols -> relu(0)=0
#pragma unroll
        for (int r = 0; r < 8; ++r) {
            float x = acc[r] + bb;
            xS[wave][g * 8 + r][col] = (f16)(x > 0.0f ? x : 0.0f);
        }
    }

    // A-fragments for layer 2 from the relu tile (K padded to 128)
    F16x16 xf[4];
#pragma unroll
    for (int kt = 0; kt < 4; ++kt)
#pragma unroll
        for (int j = 0; j < 8; ++j)
            xf[kt].u[j] = *(const unsigned*)&xS[wave][nl][kt * 32 + kform(2 * j, g)];

    for (int nt = 0; nt < 3; ++nt) {
        v8f acc = {0.f, 0.f, 0.f, 0.f, 0.f, 0.f, 0.f, 0.f};
#pragma unroll
        for (int kt = 0; kt < 4; ++kt)
            acc = __builtin_amdgcn_wmma_f32_16x16x32_f16(
                      false, xf[kt].v, false, W2[(size_t)(kt * 3 + nt) * 32 + lane],
                      (short)0, acc, false, false);
        const int col = nt * 16 + nl;
        if (col < 40) {
            const float bb = b2[col];
#pragma unroll
            for (int r = 0; r < 8; ++r)
                out[(size_t)(rowBase + g * 8 + r) * 40 + col] = acc[r] + bb;
        }
    }
}

// ---------------------------------------------------------------------------
extern "C" void kernel_launch(void* const* d_in, const int* in_sizes, int n_in,
                              void* d_out, int out_size, void* d_ws, size_t ws_size,
                              hipStream_t stream) {
    (void)in_sizes; (void)n_in; (void)out_size; (void)ws_size;
    const int*   input = (const int*)  d_in[0];
    const float* emb   = (const float*)d_in[1];
    const float* qWih  = (const float*)d_in[2];
    const float* qWhh  = (const float*)d_in[3];
    const float* qb    = (const float*)d_in[4];
    const float* oWih  = (const float*)d_in[5];
    const float* oWhh  = (const float*)d_in[6];
    const float* ob    = (const float*)d_in[7];
    const float* fc1W  = (const float*)d_in[8];
    const float* fc1b  = (const float*)d_in[9];
    const float* fc2W  = (const float*)d_in[10];
    const float* fc2b  = (const float*)d_in[11];
    float* qvals = (float*)d_out;

    char* ws = (char*)d_ws;
    f16* emb16 = (f16*)(ws + OFF_EMB);
    f16* qWp   = (f16*)(ws + OFF_QW);
    f16* oWp   = (f16*)(ws + OFF_OW);
    f16* fc1p  = (f16*)(ws + OFF_FC1);
    f16* fc2p  = (f16*)(ws + OFF_FC2);
    f16* encq  = (f16*)(ws + OFF_EQ);
    f16* enco  = (f16*)(ws + OFF_EO);

    pack_emb_kernel<<<(100 * 32 + 255) / 256, 256, 0, stream>>>(emb, emb16);
    pack_w_kernel  <<<(9 * 64 * 32 + 255) / 256, 256, 0, stream>>>(qWih, qWhh, qWp);
    pack_w_kernel  <<<(9 * 64 * 32 + 255) / 256, 256, 0, stream>>>(oWih, oWhh, oWp);
    pack_fc1_kernel<<<(16 * 7 * 32 + 255) / 256, 256, 0, stream>>>(fc1W, fc1p);
    pack_fc2_kernel<<<(4 * 3 * 32 + 255) / 256, 256, 0, stream>>>(fc2W, fc2p);

    dim3 encGrid(16384 / 32, 2);
    lstm_enc_kernel<<<encGrid, 32, 0, stream>>>(
        input, emb16, (const v16h*)qWp, qb, (const v16h*)oWp, ob, encq, enco);

    head_kernel<<<16384 / 32, 64, 0, stream>>>(
        encq, enco, (const v16h*)fc1p, fc1b, (const v16h*)fc2p, fc2b, qvals);
}
